// BiMambaBlock_35201551958097
// MI455X (gfx1250) — matrix operand order repaired and checked by reading, not compiled, not run on hardware
//
#include <hip/hip_runtime.h>

#define BATCH 2
#define SEQ 1024
#define DMODEL 1024
#define DINNER 2048
#define DSTATE 16
#define DCONV 4
#define DTRANK 64
#define NPROJ 96 /* DTRANK + 2*DSTATE */

typedef __bf16 bf16;
typedef __attribute__((ext_vector_type(16))) __bf16 bf16x16;
typedef __attribute__((ext_vector_type(8)))  __bf16 bf16x8;
typedef __attribute__((ext_vector_type(8)))  float  f32x8;

// ---------------- conversion kernels ----------------

__global__ void f32_to_bf16_kernel(const float* __restrict__ s, bf16* __restrict__ d, int n) {
  int i = blockIdx.x * blockDim.x + threadIdx.x;
  if (i < n) d[i] = (bf16)s[i];
}

// x: [B, L, DMODEL] f32 -> bf16, optionally sequence-flipped
__global__ void flip_x_bf16_kernel(const float* __restrict__ x, bf16* __restrict__ d, int flip) {
  int i = blockIdx.x * blockDim.x + threadIdx.x;
  if (i >= BATCH * SEQ * DMODEL) return;
  int k = i & (DMODEL - 1);
  int l = (i >> 10) & (SEQ - 1);
  int b = i >> 20;
  int sl = flip ? (SEQ - 1 - l) : l;
  d[i] = (bf16)x[((size_t)(b * SEQ + sl) << 10) + k];
}

// proj: [B*L, 96] -> dt bf16 [B*L, 64]
__global__ void extract_dt_kernel(const float* __restrict__ proj, bf16* __restrict__ dt) {
  int i = blockIdx.x * blockDim.x + threadIdx.x;
  if (i >= BATCH * SEQ * DTRANK) return;
  int c = i & (DTRANK - 1);
  int r = i >> 6;
  dt[i] = (bf16)proj[(size_t)r * NPROJ + c];
}

// ---------------- depthwise causal conv + SiLU ----------------
// xh lives in xz[:, 0:2048] (row stride 4096). Emits xc (f32) and xcbf (bf16).
__global__ void conv_silu_kernel(const float* __restrict__ xz,
                                 const float* __restrict__ cw,
                                 const float* __restrict__ cb,
                                 float* __restrict__ xc,
                                 bf16* __restrict__ xcbf) {
  int i = blockIdx.x * blockDim.x + threadIdx.x;
  if (i >= BATCH * SEQ * DINNER) return;
  int d = i & (DINNER - 1);
  int l = (i >> 11) & (SEQ - 1);
  int b = i >> 21;
  float v = cb[d];
#pragma unroll
  for (int k = 0; k < DCONV; ++k) {
    int ls = l - (DCONV - 1) + k;
    if (ls >= 0) v += xz[((size_t)(b * SEQ + ls)) * (2 * DINNER) + d] * cw[d * DCONV + k];
  }
  float s = v / (1.f + __expf(-v)); // silu
  xc[i] = s;
  xcbf[i] = (bf16)s;
}

// ---------------- selective scan ----------------
// One thread per (b, d) channel; 16-state h in registers; B/C broadcast in LDS.
__global__ __launch_bounds__(256) void scan_kernel(const float* __restrict__ delta, // ld 4096 (overlay in xz)
                                                   const float* __restrict__ zmat,  // xz + 2048, ld 4096
                                                   const float* __restrict__ xc,    // ld 2048
                                                   const float* __restrict__ proj,  // ld 96
                                                   const float* __restrict__ A_log,
                                                   const float* __restrict__ Dp,
                                                   bf16* __restrict__ ybf,          // [B*L, 2048]
                                                   int flip) {
  __shared__ float sB[DSTATE];
  __shared__ float sC[DSTATE];
  int b = blockIdx.x >> 3;                       // DINNER/256 = 8 blocks per batch
  int d = ((blockIdx.x & 7) << 8) + threadIdx.x;

  float Aa[DSTATE];
#pragma unroll
  for (int n = 0; n < DSTATE; ++n) Aa[n] = -__expf(A_log[d * DSTATE + n]);
  float h[DSTATE];
#pragma unroll
  for (int n = 0; n < DSTATE; ++n) h[n] = 0.f;
  float Dv = Dp[d];

  for (int l = 0; l < SEQ; ++l) {
    size_t row = (size_t)(b * SEQ + l);
    __syncthreads();
    if (threadIdx.x < 2 * DSTATE) {
      float t = proj[row * NPROJ + DTRANK + threadIdx.x];
      if (threadIdx.x < DSTATE) sB[threadIdx.x] = t;
      else                      sC[threadIdx.x - DSTATE] = t;
    }
    __syncthreads();
    float dlt = delta[row * (2 * DINNER) + d];
    float xcv = xc[row * DINNER + d];
    float du = dlt * xcv;
    float y = 0.f;
#pragma unroll
    for (int n = 0; n < DSTATE; ++n) {
      h[n] = __expf(dlt * Aa[n]) * h[n] + du * sB[n];
      y += h[n] * sC[n];
    }
    float zv = zmat[row * (2 * DINNER) + d];
    float outv = (y + xcv * Dv) * (zv / (1.f + __expf(-zv)));
    int ol = flip ? (SEQ - 1 - l) : l;
    ybf[(size_t)(b * SEQ + ol) * DINNER + d] = (bf16)outv;
  }
}

// ---------------- WMMA bf16 GEMM: C[M,N] (+)= A[M,K] @ Bw[N,K]^T ----------------
// flags bit0: epilogue softplus(v + bias[n]); bit1: accumulate into C.
// Each wave computes a (16*MT)x(16*NT) tile as MTxNT v_wmma_f32_16x16x32_bf16.
// 4 waves / block, consecutive tiles -> A-row reuse across waves via L0/L2.
template <int MT, int NT>
__global__ __launch_bounds__(128) void wmma_gemm_bf16_kernel(const bf16* __restrict__ A,
                                                             const bf16* __restrict__ Bw,
                                                             float* __restrict__ C,
                                                             const float* __restrict__ bias,
                                                             int M, int N, int K, int ldc, int flags) {
  const int lane = threadIdx.x & 31;
  const int wave = threadIdx.x >> 5;
  const int g = lane >> 4;   // half-wave
  const int ln = lane & 15;
  const int tilesN = N / (16 * NT);
  const int totalTiles = (M / (16 * MT)) * tilesN;
  const int tile = blockIdx.x * 4 + wave;   // wave-uniform
  if (tile >= totalTiles) return;           // wave-uniform exit: EXEC all-ones at WMMA
  const int tm = (tile / tilesN) * (16 * MT);
  const int tn = (tile % tilesN) * (16 * NT);

  f32x8 acc[MT][NT] = {};

  const bf16* Ar[MT];
  const bf16* Br[NT];
#pragma unroll
  for (int i = 0; i < MT; ++i) Ar[i] = A + (size_t)(tm + 16 * i + ln) * K;
#pragma unroll
  for (int j = 0; j < NT; ++j) Br[j] = Bw + (size_t)(tn + 16 * j + ln) * K;

#pragma unroll 1 // keep the live set = acc + one fragment generation (no spills)
  for (int k0 = 0; k0 < K; k0 += 32) {
    if (k0 + 64 <= K) { // prefetch next K block (gfx1250 global_prefetch path)
      __builtin_prefetch(Ar[0] + k0 + 32, 0, 0);
      __builtin_prefetch(Br[0] + k0 + 32, 0, 0);
    }
    // A fragment (this code's k order, not the ISA table order): lane(g,m): K = [16g,16g+16) contiguous, matching B
    union Frag { bf16x16 v; bf16x8 h[2]; } a[MT];
    bf16x16 b[NT];
#pragma unroll
    for (int i = 0; i < MT; ++i) {
      a[i].h[0] = *(const bf16x8*)(Ar[i] + k0 + 16 * g);
      a[i].h[1] = *(const bf16x8*)(Ar[i] + k0 + 16 * g + 8);
    }
    // B fragment: lane(g,n) holds B[k0+16g .. +15, n] = 16 contiguous bf16 of weight row n
#pragma unroll
    for (int j = 0; j < NT; ++j) b[j] = *(const bf16x16*)(Br[j] + k0 + 16 * g);

#pragma unroll
    for (int i = 0; i < MT; ++i)
#pragma unroll
      for (int j = 0; j < NT; ++j)
        acc[i][j] = __builtin_amdgcn_wmma_f32_16x16x32_bf16(false, a[i].v, false, b[j],
                                                            (short)0, acc[i][j], false, false);
  }

#pragma unroll
  for (int i = 0; i < MT; ++i)
#pragma unroll
    for (int j = 0; j < NT; ++j) {
      int cn = tn + 16 * j + ln;
      float bv = (flags & 1) ? bias[cn] : 0.f;
#pragma unroll
      for (int r = 0; r < 8; ++r) {
        float v = acc[i][j][r];
        if (flags & 1) { v += bv; v = (v > 20.f) ? v : log1pf(__expf(v)); } // softplus
        size_t idx = (size_t)(tm + 16 * i + 8 * g + r) * ldc + cn;          // C: m = r + 8g
        if (flags & 2) C[idx] += v; else C[idx] = v;
      }
    }
}

// ---------------- host launch ----------------

extern "C" void kernel_launch(void* const* d_in, const int* in_sizes, int n_in,
                              void* d_out, int out_size, void* d_ws, size_t ws_size,
                              hipStream_t stream) {
  (void)in_sizes; (void)n_in; (void)out_size; (void)ws_size;
  const float* x = (const float*)d_in[0];
  float* out = (float*)d_out;

  char* ws = (char*)d_ws;
  size_t off = 0;
  auto alloc = [&](size_t bytes) -> void* {
    void* p = ws + off;
    off = (off + bytes + 255) & ~(size_t)255;
    return p;
  };

  bf16*  xbf    = (bf16*) alloc((size_t)BATCH * SEQ * DMODEL * 2);          // 4 MB
  bf16*  w_inb  = (bf16*) alloc((size_t)2 * DINNER * DMODEL * 2);           // 8 MB
  bf16*  w_xpb  = (bf16*) alloc((size_t)NPROJ * DINNER * 2);                // 0.4 MB
  bf16*  w_dtb  = (bf16*) alloc((size_t)DINNER * DTRANK * 2);               // 0.25 MB
  bf16*  w_outb = (bf16*) alloc((size_t)DMODEL * DINNER * 2);               // 4 MB
  float* xz     = (float*)alloc((size_t)BATCH * SEQ * 2 * DINNER * 4);      // 32 MB (xh|z; delta overlays xh)
  float* xc     = (float*)alloc((size_t)BATCH * SEQ * DINNER * 4);          // 16 MB
  bf16*  xcbf   = (bf16*) alloc((size_t)BATCH * SEQ * DINNER * 2);          // 8 MB (reused as ybf)
  float* proj   = (float*)alloc((size_t)BATCH * SEQ * NPROJ * 4);           // 0.75 MB
  bf16*  dtbf   = (bf16*) alloc((size_t)BATCH * SEQ * DTRANK * 2);          // 0.25 MB
  bf16*  ybf    = xcbf; // xcbf dead after x-proj GEMM

  const int ROWS = BATCH * SEQ; // 2048

  for (int p = 0; p < 2; ++p) {
    const float* in_w    = (const float*)d_in[1 + 9 * p + 0];
    const float* conv_w  = (const float*)d_in[1 + 9 * p + 1];
    const float* conv_b  = (const float*)d_in[1 + 9 * p + 2];
    const float* xproj_w = (const float*)d_in[1 + 9 * p + 3];
    const float* dt_w    = (const float*)d_in[1 + 9 * p + 4];
    const float* dt_b    = (const float*)d_in[1 + 9 * p + 5];
    const float* A_log   = (const float*)d_in[1 + 9 * p + 6];
    const float* Dp      = (const float*)d_in[1 + 9 * p + 7];
    const float* out_w   = (const float*)d_in[1 + 9 * p + 8];

    // bf16 staging (flip x for the backward direction)
    flip_x_bf16_kernel<<<(BATCH * SEQ * DMODEL + 255) / 256, 256, 0, stream>>>(x, xbf, p);
    f32_to_bf16_kernel<<<(2 * DINNER * DMODEL + 255) / 256, 256, 0, stream>>>(in_w, w_inb, 2 * DINNER * DMODEL);
    f32_to_bf16_kernel<<<(NPROJ * DINNER + 255) / 256, 256, 0, stream>>>(xproj_w, w_xpb, NPROJ * DINNER);
    f32_to_bf16_kernel<<<(DINNER * DTRANK + 255) / 256, 256, 0, stream>>>(dt_w, w_dtb, DINNER * DTRANK);
    f32_to_bf16_kernel<<<(DMODEL * DINNER + 255) / 256, 256, 0, stream>>>(out_w, w_outb, DMODEL * DINNER);

    // in-proj: xz[2048,4096] = xbf @ in_w^T (K=1024), 64x64 wave tiles
    { int tiles = (ROWS / 64) * (2 * DINNER / 64);
      wmma_gemm_bf16_kernel<4, 4><<<(tiles + 3) / 4, 128, 0, stream>>>(xbf, w_inb, xz, nullptr,
                                                                       ROWS, 2 * DINNER, DMODEL, 2 * DINNER, 0); }

    // depthwise conv + silu on xh half
    conv_silu_kernel<<<(BATCH * SEQ * DINNER + 255) / 256, 256, 0, stream>>>(xz, conv_w, conv_b, xc, xcbf);

    // x-proj: proj[2048,96] = xcbf @ xproj_w^T (K=2048), 32x32 wave tiles (N=96)
    { int tiles = (ROWS / 32) * (NPROJ / 32);
      wmma_gemm_bf16_kernel<2, 2><<<(tiles + 3) / 4, 128, 0, stream>>>(xcbf, w_xpb, proj, nullptr,
                                                                       ROWS, NPROJ, DINNER, NPROJ, 0); }

    extract_dt_kernel<<<(ROWS * DTRANK + 255) / 256, 256, 0, stream>>>(proj, dtbf);

    // dt-proj with fused bias+softplus: delta overlays xh half of xz (ldc=4096)
    { int tiles = (ROWS / 64) * (DINNER / 64);
      wmma_gemm_bf16_kernel<4, 4><<<(tiles + 3) / 4, 128, 0, stream>>>(dtbf, w_dtb, xz, dt_b,
                                                                       ROWS, DINNER, DTRANK, 2 * DINNER, 1); }

    // selective scan -> ybf (bf16), sequence un-flipped on write for backward pass
    scan_kernel<<<BATCH * (DINNER / 256), 256, 0, stream>>>(xz, xz + DINNER, xc, proj, A_log, Dp, ybf, p);

    // out-proj: d_out (+)= ybf @ out_w^T  (beta=0 fwd, beta=1 bwd)
    { int tiles = (ROWS / 64) * (DMODEL / 64);
      wmma_gemm_bf16_kernel<4, 4><<<(tiles + 3) / 4, 128, 0, stream>>>(ybf, w_outb, out, nullptr,
                                                                       ROWS, DMODEL, DINNER, DMODEL, p ? 2 : 0); }
  }
}
